// DANNModel_59768764891410
// MI455X (gfx1250) — matrix operand order repaired and checked by reading, not compiled, not run on hardware
//
#include <hip/hip_runtime.h>
#include <hip/hip_bf16.h>
#include <math.h>

typedef __attribute__((ext_vector_type(16))) __bf16 v16bf;
typedef __attribute__((ext_vector_type(8)))  __bf16 v8bf;
typedef __attribute__((ext_vector_type(8)))  float  v8f;

#define B_SZ   128
#define S_LEN  512
#define F_INPUT 64
#define H_DIM  512
#define K_DIM  576                 // 64 (x) + 512 (h)
#define N_COLS 2053                // 4H + 5
#define N_PAD  2064                // padded to N-tile multiple of 16
#define N_WG   32                  // j-blocks of 16 hidden units
#define EPSF   1e-5f

// LDS B-slice geometry: 80 rows (4 gate tiles + d tile), padded row stride
#define LROWS      80
#define LSTRIDE_E  584             // elements per row (584*2 = 1168 B = 292 DW, bank-friendly)
#define LCHUNKS    72              // 16-byte chunks of real data per row (576*2/16)

// ---- workspace layout (bytes, all 256-aligned) ----
#define O_CNT   0
#define O_HBF   256                                       // bf16 h, double buffered [2][128][512]
#define O_XBF   (O_HBF  + 2*B_SZ*H_DIM*2)                 // bf16 x [128][512][64]
#define O_BT    (O_XBF  + B_SZ*S_LEN*F_INPUT*2)           // bf16 Bt [2064][576]
#define O_HOUT  (O_BT   + N_PAD*K_DIM*2)                  // f32 h last [128][512]
#define O_COUT  (O_HOUT + B_SZ*H_DIM*4)                   // f32 c1..c5 [5][128][512]
#define O_YB    (O_COUT + 5*B_SZ*H_DIM*4)                 // f32 head hidden [6][128][256]

__device__ __forceinline__ float sigm(float x) { return 1.0f / (1.0f + __expf(-x)); }

// ---------------- init kernels ----------------
__global__ void init_state(__bf16* __restrict__ h_bf, unsigned int* __restrict__ bar) {
    const int i = blockIdx.x * 256 + threadIdx.x;
    if (i < 2 * B_SZ * H_DIM) h_bf[i] = (__bf16)0.0f;
    if (i == 0) *bar = 0u;
}

__global__ void cvt_x(const float* __restrict__ x, __bf16* __restrict__ xb, int n) {
    const int i = blockIdx.x * 256 + threadIdx.x;
    if (i < n) xb[i] = (__bf16)x[i];
}

// Bt[n][k] = (n<2053) ? (k<64 ? W[k][n] : U[k-64][n]) : 0   (transposed + padded, bf16)
__global__ void build_bt(const float* __restrict__ W, const float* __restrict__ U,
                         __bf16* __restrict__ Bt) {
    const int i = blockIdx.x * 256 + threadIdx.x;
    if (i >= N_PAD * K_DIM) return;
    const int n = i / K_DIM;
    const int k = i - n * K_DIM;
    float v = 0.0f;
    if (n < N_COLS)
        v = (k < F_INPUT) ? W[(size_t)k * N_COLS + n] : U[(size_t)(k - F_INPUT) * N_COLS + n];
    Bt[i] = (__bf16)v;
}

// ---------------- persistent recurrence kernel ----------------
__global__ __launch_bounds__(256) void mclstm_recurrence(
    const float* __restrict__ bias, const float* __restrict__ bias1,
    const float* __restrict__ kernel1,
    const __bf16* __restrict__ x_bf, const __bf16* __restrict__ Bt,
    __bf16* __restrict__ h_bf,            // [2][128][512]
    float* __restrict__ h_out, float* __restrict__ c_out,   // c_out: [5][128][512]
    float* __restrict__ d_out_d,          // [128][5]
    unsigned int* __restrict__ bar)
{
    __shared__ __bf16 bmat[LROWS * LSTRIDE_E];   // ~93 KB: this WG's B slice, WMMA-ready
    __shared__ float  draw[B_SZ][8];             // raw d-gate values
    __shared__ float  dsm[B_SZ][8];              // softmax(tanh(d-gate))

    const int tid  = threadIdx.x;
    const int lane = tid & 31;
    const int wave = tid >> 5;        // 0..7 -> M-tile (16 batch rows)
    const int l15  = lane & 15;
    const int hi16 = lane >> 4;
    const int j0   = blockIdx.x * 16; // this WG's hidden-unit block
    const int j    = j0 + l15;        // one hidden unit per lane (mod 16)

    // ---- stage B slice into LDS (once; constant across all 512 steps) ----
    // local row rl = g*16 + t  ->  global row n = g*512 + j0 + t (g<4), 2048 + t (g==4)
    for (int p = tid; p < LROWS * LCHUNKS; p += 256) {
        const int rl = p / LCHUNKS;
        const int ch = p - rl * LCHUNKS;
        const int g  = rl >> 4;
        const int t  = rl & 15;
        const int n  = (g < 4) ? (g * H_DIM + j0 + t) : (2048 + t);
        const uint4 v = *((const uint4*)(Bt + (size_t)n * K_DIM) + ch);
        *((uint4*)(bmat + (size_t)rl * LSTRIDE_E) + ch) = v;
    }

    // per-lane constants for hidden unit j
    const float s1 = bias1[j];
    const float s2 = bias1[H_DIM + j];
    const float kk0 = kernel1[0 * H_DIM + j], kk1 = kernel1[1 * H_DIM + j],
                kk2 = kernel1[2 * H_DIM + j], kk3 = kernel1[3 * H_DIM + j],
                kk4 = kernel1[4 * H_DIM + j];
    const float bg0 = bias[0 * H_DIM + j], bg1 = bias[1 * H_DIM + j],
                bg2 = bias[2 * H_DIM + j], bg3 = bias[3 * H_DIM + j];
    const float bd = (l15 < 5) ? bias[4 * H_DIM + l15] : 0.0f;

    // LDS B-fragment row pointers: lane = N column, 16 contiguous K at hi16*16
    const __bf16* lb0 = bmat + (size_t)(0 * 16 + l15) * LSTRIDE_E + hi16 * 16;
    const __bf16* lb1 = bmat + (size_t)(1 * 16 + l15) * LSTRIDE_E + hi16 * 16;
    const __bf16* lb2 = bmat + (size_t)(2 * 16 + l15) * LSTRIDE_E + hi16 * 16;
    const __bf16* lb3 = bmat + (size_t)(3 * 16 + l15) * LSTRIDE_E + hi16 * 16;
    const __bf16* lb4 = bmat + (size_t)(4 * 16 + l15) * LSTRIDE_E + hi16 * 16;

    const int rowA = wave * 16 + l15;                       // A-row (batch) for this lane
    const __bf16* xrow = x_bf + (size_t)rowA * S_LEN * F_INPUT + hi16 * 16;

    float cst[8];                                           // cell state c, in registers
    #pragma unroll
    for (int r = 0; r < 8; ++r) cst[r] = 0.0f;

    __syncthreads();                                        // B slice visible to all waves

    for (int s = 0; s < S_LEN; ++s) {
        const __bf16* hr = h_bf + (size_t)(s & 1) * (B_SZ * H_DIM);
        __bf16*       hw = h_bf + (size_t)((s + 1) & 1) * (B_SZ * H_DIM);

        v8f a0 = {}, a1 = {}, a2 = {}, a3 = {}, a4 = {};
        const __bf16* ax = xrow + (size_t)s * F_INPUT;
        const __bf16* ah = hr + (size_t)rowA * H_DIM + hi16 * 16;

        #pragma unroll
        for (int kt = 0; kt < 18; ++kt) {
            const __bf16* ap = (kt < 2) ? (ax + kt * 32) : (ah + (kt - 2) * 32);
            v8bf alo = *(const v8bf*)ap;            // K sub-block 0..7 (lanes 0-15) / 16..23
            v8bf ahi = *(const v8bf*)(ap + 8);      // K sub-block 8..15 / 24..31
            v16bf a = __builtin_shufflevector(alo, ahi,
                       0,1,2,3,4,5,6,7,8,9,10,11,12,13,14,15);
            v16bf b0 = *(const v16bf*)(lb0 + kt * 32);
            v16bf b1 = *(const v16bf*)(lb1 + kt * 32);
            v16bf b2 = *(const v16bf*)(lb2 + kt * 32);
            v16bf b3 = *(const v16bf*)(lb3 + kt * 32);
            v16bf b4 = *(const v16bf*)(lb4 + kt * 32);
            a0 = __builtin_amdgcn_wmma_f32_16x16x32_bf16(false, a, false, b0, (short)0, a0, false, false);
            a1 = __builtin_amdgcn_wmma_f32_16x16x32_bf16(false, a, false, b1, (short)0, a1, false, false);
            a2 = __builtin_amdgcn_wmma_f32_16x16x32_bf16(false, a, false, b2, (short)0, a2, false, false);
            a3 = __builtin_amdgcn_wmma_f32_16x16x32_bf16(false, a, false, b3, (short)0, a3, false, false);
            a4 = __builtin_amdgcn_wmma_f32_16x16x32_bf16(false, a, false, b4, (short)0, a4, false, false);
        }

        if (s + 1 < S_LEN) __builtin_prefetch(ax + F_INPUT, 0, 1);   // next step's x_t

        // d-gate: cross-lane softmax via small LDS buffer
        if (l15 < 5) {
            #pragma unroll
            for (int r = 0; r < 8; ++r)
                draw[wave * 16 + hi16 * 8 + r][l15] = a4[r] + bd;
        }
        __syncthreads();
        if (tid < B_SZ) {
            float v0 = tanhf(draw[tid][0]), v1 = tanhf(draw[tid][1]),
                  v2 = tanhf(draw[tid][2]), v3 = tanhf(draw[tid][3]),
                  v4 = tanhf(draw[tid][4]);
            float mx = fmaxf(fmaxf(fmaxf(v0, v1), fmaxf(v2, v3)), v4);
            v0 = __expf(v0 - mx); v1 = __expf(v1 - mx); v2 = __expf(v2 - mx);
            v3 = __expf(v3 - mx); v4 = __expf(v4 - mx);
            const float inv = 1.0f / (v0 + v1 + v2 + v3 + v4);
            dsm[tid][0] = v0 * inv; dsm[tid][1] = v1 * inv; dsm[tid][2] = v2 * inv;
            dsm[tid][3] = v3 * inv; dsm[tid][4] = v4 * inv;
        }
        __syncthreads();

        // cell update fully in registers (i,f,g,o share the C-tile layout per lane)
        const bool last = (s == S_LEN - 1);
        #pragma unroll
        for (int r = 0; r < 8; ++r) {
            const int row = wave * 16 + hi16 * 8 + r;
            const float gi = sigm(a0[r] + bg0);
            const float gf = sigm(a1[r] + bg1);
            const float gg = tanhf(a2[r] + bg2);
            const float go = sigm(a3[r] + bg3);
            const float c5v = cst[r];
            const float c1v = tanhf(gg);
            const float c3v = gf * c5v + gi * gg;
            const float c2v = s2 * c3v + (1.0f - s2) * c1v;
            const float c4v = s1 * c3v + (1.0f - s1) * c5v;
            const float cn = dsm[row][0] * kk0 * c1v + dsm[row][1] * kk1 * c2v
                           + dsm[row][2] * kk2 * c3v + dsm[row][3] * kk3 * c4v
                           + dsm[row][4] * kk4 * c5v;
            const float hn = go * tanhf(cn);
            cst[r] = cn;
            const size_t idx = (size_t)row * H_DIM + j;
            hw[idx] = (__bf16)hn;
            if (last) {
                h_out[idx] = hn;
                c_out[0 * (size_t)(B_SZ * H_DIM) + idx] = c1v;
                c_out[1 * (size_t)(B_SZ * H_DIM) + idx] = c2v;
                c_out[2 * (size_t)(B_SZ * H_DIM) + idx] = c3v;
                c_out[3 * (size_t)(B_SZ * H_DIM) + idx] = c4v;
                c_out[4 * (size_t)(B_SZ * H_DIM) + idx] = c5v;
            }
        }
        if (last && blockIdx.x == 0 && tid < B_SZ) {
            #pragma unroll
            for (int p = 0; p < 5; ++p) d_out_d[tid * 5 + p] = dsm[tid][p];
        }

        // device-wide step barrier (release h_new, acquire all partitions)
        __syncthreads();
        __threadfence();
        if (tid == 0) {
            __hip_atomic_fetch_add(bar, 1u, __ATOMIC_ACQ_REL, __HIP_MEMORY_SCOPE_AGENT);
            const unsigned target = (unsigned)(s + 1) * (unsigned)gridDim.x;
            while (__hip_atomic_load(bar, __ATOMIC_ACQUIRE, __HIP_MEMORY_SCOPE_AGENT) < target)
                __builtin_amdgcn_s_sleep(1);
        }
        __syncthreads();
        __threadfence();
    }
}

// ---------------- head kernels (0.3% of FLOPs, plain f32) ----------------
// block = (feat 0..5) x (col-group of 32): feat0 = h@rw1, feat1..5 = c_i@dw1, relu'd
__global__ __launch_bounds__(256) void heads_gemm1(
    const float* __restrict__ h_out, const float* __restrict__ c_out,
    const float* __restrict__ rw1, const float* __restrict__ rb1,
    const float* __restrict__ dw1, const float* __restrict__ db1,
    float* __restrict__ yb)
{
    const int f  = blockIdx.x >> 3;
    const int cg = blockIdx.x & 7;
    const float* feat = (f == 0) ? h_out : (c_out + (size_t)(f - 1) * B_SZ * H_DIM);
    const float* w    = (f == 0) ? rw1 : dw1;
    const float* bv   = (f == 0) ? rb1 : db1;
    float* out = yb + (size_t)f * B_SZ * 256;
    for (int p = threadIdx.x; p < B_SZ * 32; p += 256) {
        const int row = p >> 5;
        const int col = cg * 32 + (p & 31);
        const float* fr = feat + (size_t)row * H_DIM;
        float acc = bv[col];
        for (int k2 = 0; k2 < H_DIM; ++k2) acc += fr[k2] * w[(size_t)k2 * 256 + col];
        out[(size_t)row * 256 + col] = fmaxf(acc, 0.0f);
    }
}

// block 0: rul path; blocks 1..5: dom(c_i) with batchnorm + log_softmax
__global__ __launch_bounds__(256) void heads_final(
    float* __restrict__ yb,
    const float* __restrict__ rw2, const float* __restrict__ rb2,
    const float* __restrict__ rbng, const float* __restrict__ rbnb,
    const float* __restrict__ rw3, const float* __restrict__ rb3,
    const float* __restrict__ dbng, const float* __restrict__ dbnb,
    const float* __restrict__ dw2, const float* __restrict__ db2,
    float* __restrict__ out)
{
    __shared__ float r2[B_SZ][128];
    const int t = threadIdx.x;
    if (blockIdx.x == 0) {
        const float* r1 = yb;   // feat 0
        for (int p = t; p < B_SZ * 128; p += 256) {
            const int row = p >> 7, col = p & 127;
            const float* rr = r1 + (size_t)row * 256;
            float acc = rb2[col];
            for (int k2 = 0; k2 < 256; ++k2) acc += rr[k2] * rw2[(size_t)k2 * 128 + col];
            r2[row][col] = fmaxf(acc, 0.0f);
        }
        __syncthreads();
        if (t < 128) {
            float m = 0.0f, sq = 0.0f;
            for (int r = 0; r < B_SZ; ++r) { const float x = r2[r][t]; m += x; sq += x * x; }
            m *= (1.0f / B_SZ);
            const float var = sq * (1.0f / B_SZ) - m * m;
            const float inv = rsqrtf(var + EPSF);
            const float g = rbng[t], b = rbnb[t];
            for (int r = 0; r < B_SZ; ++r) r2[r][t] = (r2[r][t] - m) * inv * g + b;
        }
        __syncthreads();
        if (t < B_SZ) {
            float acc = rb3[0];
            for (int c = 0; c < 128; ++c) acc += r2[t][c] * rw3[c];
            out[t] = acc;
        }
    } else {
        const int f = blockIdx.x;                 // 1..5
        float* y = yb + (size_t)f * B_SZ * 256;
        {
            const int col = t;                    // 256 cols, one per thread
            float m = 0.0f, sq = 0.0f;
            for (int r = 0; r < B_SZ; ++r) { const float x = y[(size_t)r * 256 + col]; m += x; sq += x * x; }
            m *= (1.0f / B_SZ);
            const float var = sq * (1.0f / B_SZ) - m * m;
            const float inv = rsqrtf(var + EPSF);
            const float g = dbng[col], b = dbnb[col];
            for (int r = 0; r < B_SZ; ++r) {
                const float x = y[(size_t)r * 256 + col];
                y[(size_t)r * 256 + col] = (x - m) * inv * g + b;
            }
        }
        __threadfence_block();
        __syncthreads();
        if (t < B_SZ) {
            const float* yr = y + (size_t)t * 256;
            float z0 = db2[0], z1 = db2[1];
            for (int c = 0; c < 256; ++c) { const float x = yr[c]; z0 += x * dw2[c * 2 + 0]; z1 += x * dw2[c * 2 + 1]; }
            const float mx = fmaxf(z0, z1);
            const float ls = mx + logf(__expf(z0 - mx) + __expf(z1 - mx));
            out[B_SZ + (size_t)(f - 1) * 256 + t * 2 + 0] = z0 - ls;
            out[B_SZ + (size_t)(f - 1) * 256 + t * 2 + 1] = z1 - ls;
        }
    }
}

extern "C" void kernel_launch(void* const* d_in, const int* in_sizes, int n_in,
                              void* d_out, int out_size, void* d_ws, size_t ws_size,
                              hipStream_t stream)
{
    (void)in_sizes; (void)n_in; (void)out_size; (void)ws_size;
    const float* input_data = (const float*)d_in[0];
    const float* W       = (const float*)d_in[1];
    const float* U       = (const float*)d_in[2];
    const float* kernel1 = (const float*)d_in[3];
    const float* bias    = (const float*)d_in[4];
    const float* bias1   = (const float*)d_in[5];
    const float* rw1  = (const float*)d_in[6];
    const float* rb1  = (const float*)d_in[7];
    const float* rw2  = (const float*)d_in[8];
    const float* rb2  = (const float*)d_in[9];
    const float* rbng = (const float*)d_in[10];
    const float* rbnb = (const float*)d_in[11];
    const float* rw3  = (const float*)d_in[12];
    const float* rb3  = (const float*)d_in[13];
    const float* dw1  = (const float*)d_in[14];
    const float* db1  = (const float*)d_in[15];
    const float* dbng = (const float*)d_in[16];
    const float* dbnb = (const float*)d_in[17];
    const float* dw2  = (const float*)d_in[18];
    const float* db2  = (const float*)d_in[19];

    char* ws = (char*)d_ws;
    unsigned int* bar = (unsigned int*)(ws + O_CNT);
    __bf16* h_bf = (__bf16*)(ws + O_HBF);
    __bf16* x_bf = (__bf16*)(ws + O_XBF);
    __bf16* Bt   = (__bf16*)(ws + O_BT);
    float* h_out = (float*)(ws + O_HOUT);
    float* c_out = (float*)(ws + O_COUT);
    float* yb    = (float*)(ws + O_YB);
    float* out   = (float*)d_out;

    init_state<<<(2 * B_SZ * H_DIM + 255) / 256, 256, 0, stream>>>(h_bf, bar);
    cvt_x<<<(B_SZ * S_LEN * F_INPUT + 255) / 256, 256, 0, stream>>>(
        input_data, x_bf, B_SZ * S_LEN * F_INPUT);
    build_bt<<<(N_PAD * K_DIM + 255) / 256, 256, 0, stream>>>(W, U, Bt);
    mclstm_recurrence<<<N_WG, 256, 0, stream>>>(
        bias, bias1, kernel1, x_bf, Bt, h_bf, h_out, c_out,
        out + 128 + 5 * 256, bar);
    heads_gemm1<<<48, 256, 0, stream>>>(h_out, c_out, rw1, rb1, dw1, db1, yb);
    heads_final<<<6, 256, 0, stream>>>(yb, rw2, rb2, rbng, rbnb, rw3, rb3,
                                       dbng, dbnb, dw2, db2, out);
}
